// NlsConv2d_31224412242787
// MI455X (gfx1250) — matrix operand order repaired and checked by reading, not compiled, not run on hardware
//
#include <hip/hip_runtime.h>

// ---------------- problem constants (match reference setup_inputs) ----------
#define B_  8
#define C_  64
#define O_  64
#define H_  160
#define W_  160
#define HW_ (H_ * W_)
#define K_  9      // deform taps / top-k
#define WS_ 9      // search window side

typedef __attribute__((ext_vector_type(16))) _Float16 v16h;
typedef __attribute__((ext_vector_type(8)))  _Float16 v8h;
typedef __attribute__((ext_vector_type(2)))  _Float16 v2h;
typedef __attribute__((ext_vector_type(8)))  float    v8f;

// ---------------------------------------------------------------------------
// Kernel 0: reg_w [O][C][3][3] f32  ->  regw_h [l][o][c] f16  (l = ky*3+kx)
// ---------------------------------------------------------------------------
__global__ __launch_bounds__(256) void k_convert_regw(const float* __restrict__ rw,
                                                      _Float16* __restrict__ out) {
    int i = blockIdx.x * 256 + threadIdx.x;           // 9*64*64 = 36864
    if (i >= K_ * O_ * C_) return;
    int l = i >> 12;            // /4096
    int o = (i >> 6) & 63;
    int c = i & 63;
    out[i] = (_Float16)rw[(o * C_ + c) * 9 + l];
}

// ---------------------------------------------------------------------------
// Kernel 1: fused 3x3 conv (2 offset channels + 9 mod channels), zero pad.
//   soff[(b*2+ch)*HW + pix] = clip(conv, -15, 15)
//   mbuf[(b*9+l)*HW + pix]  = 2*sigmoid(conv)
// ---------------------------------------------------------------------------
__global__ __launch_bounds__(256) void k_small_conv(const float* __restrict__ x,
                                                    const float* __restrict__ ow,
                                                    const float* __restrict__ ob,
                                                    const float* __restrict__ mw,
                                                    const float* __restrict__ mb,
                                                    float* __restrict__ soff,
                                                    float* __restrict__ mbuf) {
    __shared__ float wl[11 * C_ * 9];                 // [oc][c][k], 25344 B
    int tid = threadIdx.x;
    for (int i = tid; i < 11 * C_ * 9; i += 256)
        wl[i] = (i < 2 * C_ * 9) ? ow[i] : mw[i - 2 * C_ * 9];
    __syncthreads();

    int pg  = blockIdx.x * 256 + tid;                 // B*HW = 204800
    int b   = pg / HW_;
    int pix = pg % HW_;
    int h   = pix / W_;
    int w   = pix % W_;

    float acc[11];
    acc[0] = ob[0]; acc[1] = ob[1];
#pragma unroll
    for (int l = 0; l < 9; ++l) acc[2 + l] = mb[l];

    const float* xb = x + (size_t)b * C_ * HW_;
#pragma unroll
    for (int k = 0; k < 9; ++k) {
        int yy = h + k / 3 - 1, xx = w + k % 3 - 1;
        if (yy < 0 || yy >= H_ || xx < 0 || xx >= W_) continue;
        int base = yy * W_ + xx;
        for (int c = 0; c < C_; ++c) {
            float v = xb[(size_t)c * HW_ + base];
#pragma unroll
            for (int oc = 0; oc < 11; ++oc)
                acc[oc] = fmaf(v, wl[(oc * C_ + c) * 9 + k], acc[oc]);
        }
    }

    soff[((size_t)b * 2 + 0) * HW_ + pix] = fminf(fmaxf(acc[0], -15.f), 15.f);
    soff[((size_t)b * 2 + 1) * HW_ + pix] = fminf(fmaxf(acc[1], -15.f), 15.f);
#pragma unroll
    for (int l = 0; l < 9; ++l)
        mbuf[((size_t)b * 9 + l) * HW_ + pix] = 2.f / (1.f + expf(-acc[2 + l]));
}

// ---------------------------------------------------------------------------
// Kernel 2: non-local search (81 windows, channel dot), stable top-9,
// then emit per-tap sample coordinates replicating the reference's x/y flip:
//   py_tap = h + (l/3 - 1) + (kx - w)   ;  px_tap = w + (l%3 - 1) + (ky - h)
// One pixel per thread; center vector staged in LDS (f32).
// ---------------------------------------------------------------------------
__global__ __launch_bounds__(128) void k_search(const float* __restrict__ x,
                                                const float* __restrict__ soff,
                                                float* __restrict__ coords) {
    __shared__ float ctr[C_ * 128];                   // 32 KB
    int tid = threadIdx.x;
    int pg  = blockIdx.x * 128 + tid;                 // 1600 blocks * 128
    int b   = pg / HW_;
    int pix = pg % HW_;
    int h   = pix / W_;
    int w   = pix % W_;

    const float* xb = x + (size_t)b * C_ * HW_;
    for (int c = 0; c < C_; ++c)                      // coalesced across lanes
        ctr[c * 128 + tid] = xb[(size_t)c * HW_ + pix];
    // each thread only reads its own column; no barrier required

    float cy = (float)h + soff[((size_t)b * 2 + 0) * HW_ + pix];
    float cx = (float)w + soff[((size_t)b * 2 + 1) * HW_ + pix];

    float bestv[K_]; int besti[K_];
#pragma unroll
    for (int i = 0; i < K_; ++i) { bestv[i] = -3.4e38f; besti[i] = 0; }

    for (int t = 0; t < WS_ * WS_; ++t) {
        float wy = (float)(t / WS_ - 4);
        float wx = (float)(t % WS_ - 4);
        float py = fminf(fmaxf(cy + wy, 0.f), (float)(H_ - 1));
        float px = fminf(fmaxf(cx + wx, 0.f), (float)(W_ - 1));
        float y0f = floorf(py), x0f = floorf(px);
        float fy = py - y0f, fx = px - x0f;
        int y0 = (int)y0f, x0 = (int)x0f;
        int y1 = min(y0 + 1, H_ - 1), x1 = min(x0 + 1, W_ - 1);
        // after clip the only OOB tap has exactly zero weight -> clamp is safe
        float w00 = (1.f - fy) * (1.f - fx), w01 = (1.f - fy) * fx;
        float w10 = fy * (1.f - fx),         w11 = fy * fx;
        int o00 = y0 * W_ + x0, o01 = y0 * W_ + x1;
        int o10 = y1 * W_ + x0, o11 = y1 * W_ + x1;

        float d = 0.f;
        for (int c = 0; c < C_; ++c) {
            const float* xc = xb + (size_t)c * HW_;
            float s = w00 * xc[o00] + w01 * xc[o01] + w10 * xc[o10] + w11 * xc[o11];
            d = fmaf(ctr[c * 128 + tid], s, d);
        }

        if (d > bestv[K_ - 1]) {                      // stable insertion (ties: lower idx first)
            float nv[K_]; int ni[K_];
#pragma unroll
            for (int j = 0; j < K_; ++j) {
                bool a  = d > bestv[j];
                bool sh = (j > 0) && (d > bestv[j - 1]);
                nv[j] = a ? (sh ? bestv[j - 1] : d) : bestv[j];
                ni[j] = a ? (sh ? besti[j - 1] : t) : besti[j];
            }
#pragma unroll
            for (int j = 0; j < K_; ++j) { bestv[j] = nv[j]; besti[j] = ni[j]; }
        }
    }

#pragma unroll
    for (int l = 0; l < K_; ++l) {
        int t = besti[l];
        float ky = fminf(fmaxf(cy + (float)(t / WS_ - 4), 0.f), (float)(H_ - 1));
        float kx = fminf(fmaxf(cx + (float)(t % WS_ - 4), 0.f), (float)(W_ - 1));
        float pyt = (float)(h + l / 3 - 1) + (kx - (float)w);  // reference flip
        float pxt = (float)(w + l % 3 - 1) + (ky - (float)h);
        size_t ci = ((size_t)pg * K_ + l) * 2;
        coords[ci]     = pyt;
        coords[ci + 1] = pxt;
    }
}

// ---------------------------------------------------------------------------
// Kernel 3: modulated deformable conv via WMMA f16->f32.
// Block: 64 pixels (flattened) x all 64 outputs, 256 threads (8 waves).
// Per tap: sample S[p][c] = m_l * zero-pad-bilerp(x)  (f16, LDS pitch 72),
// packed channel-pair b32 LDS stores; zero-weight fast path skips gathers.
// Then 8 waves each do 2 (16x16) D-subtiles, K=64 -> 2x v_wmma_f32_16x16x32_f16.
// ---------------------------------------------------------------------------
#define SPITCH 72   // halves; row = 144 B (16B aligned; stride 36 banks, gcd 4 -> conflict-free b128)

__global__ __launch_bounds__(256) void k_deform_wmma(const float* __restrict__ x,
                                                     const float* __restrict__ coords,
                                                     const float* __restrict__ mbuf,
                                                     const _Float16* __restrict__ regw,
                                                     const float* __restrict__ regb,
                                                     float* __restrict__ out) {
    __shared__ __align__(16) _Float16 S[64 * SPITCH];

    int tid  = threadIdx.x;
    int blk  = blockIdx.x;                 // b*(HW/64) + tile
    int b    = blk / (HW_ / 64);
    int pix0 = (blk % (HW_ / 64)) * 64;
    int lane = tid & 31;
    int wave = tid >> 5;

    // WMMA subtile assignment: wave -> o-tile = wave/2, p-tiles = (wave&1)*2 + {0,1}
    int ot    = wave >> 1;
    int pt0   = (wave & 1) * 2;
    int obase = ot * 16 + ((lane >> 4) << 3);          // C/D layout: o = obase + r

    v8f acc0, acc1;
#pragma unroll
    for (int r = 0; r < 8; ++r) { acc0[r] = regb[obase + r]; acc1[r] = acc0[r]; }

    int p  = tid & 63;                     // sampling: fixed pixel per thread
    int g  = (tid >> 6) << 1;              // channel-pair base: 0,2,4,6
    const float* xb = x + (size_t)b * C_ * HW_;

    for (int l = 0; l < K_; ++l) {
        // ---- sampling stage: zero-padding bilerp, modulation folded into weights
        size_t ci = (((size_t)b * HW_ + pix0 + p) * K_ + l) * 2;
        float py = coords[ci], px = coords[ci + 1];
        float ml = mbuf[((size_t)b * K_ + l) * HW_ + pix0 + p];

        float y0f = floorf(py), x0f = floorf(px);
        float fy = py - y0f, fx = px - x0f;
        int y0 = (int)y0f, x0 = (int)x0f;
        float wy0 = (y0 >= 0     && y0 <= H_ - 1) ? (1.f - fy) : 0.f;
        float wy1 = (y0 + 1 >= 0 && y0 + 1 <= H_ - 1) ? fy : 0.f;
        float wx0 = (x0 >= 0     && x0 <= W_ - 1) ? (1.f - fx) : 0.f;
        float wx1 = (x0 + 1 >= 0 && x0 + 1 <= W_ - 1) ? fx : 0.f;
        float w00 = wy0 * wx0 * ml, w01 = wy0 * wx1 * ml;
        float w10 = wy1 * wx0 * ml, w11 = wy1 * wx1 * ml;
        int iy0 = min(max(y0, 0), H_ - 1),     iy1 = min(max(y0 + 1, 0), H_ - 1);
        int ix0 = min(max(x0, 0), W_ - 1),     ix1 = min(max(x0 + 1, 0), W_ - 1);
        int o00 = iy0 * W_ + ix0, o01 = iy0 * W_ + ix1;
        int o10 = iy1 * W_ + ix0, o11 = iy1 * W_ + ix1;

        // weights are all >= 0 (validity x modulation folded in): sum==0 -> all zero
        if ((w00 + w01 + w10 + w11) != 0.f) {
#pragma unroll
            for (int i = 0; i < 8; ++i) {
                int c = g + (i << 3);
                const float* xc0 = xb + (size_t)c * HW_;
                const float* xc1 = xc0 + HW_;
                float s0 = w00 * xc0[o00] + w01 * xc0[o01] + w10 * xc0[o10] + w11 * xc0[o11];
                float s1 = w00 * xc1[o00] + w01 * xc1[o01] + w10 * xc1[o10] + w11 * xc1[o11];
                v2h pk; pk[0] = (_Float16)s0; pk[1] = (_Float16)s1;
                *(v2h*)&S[p * SPITCH + c] = pk;        // packed b32 LDS store
            }
        } else {
            v2h z; z[0] = (_Float16)0.f; z[1] = (_Float16)0.f;
#pragma unroll
            for (int i = 0; i < 8; ++i)
                *(v2h*)&S[p * SPITCH + g + (i << 3)] = z;
        }
        __syncthreads();

        // ---- WMMA stage (all 256 threads active: EXEC all-ones as required)
#pragma unroll
        for (int kc = 0; kc < 2; ++kc) {
            // A 16x32 f16: lane L -> row o=ot*16+(L%16), K = kc*32 + (L/16)*16 + {0..15}
            const _Float16* ab = regw
                + (size_t)((l * O_ + ot * 16 + (lane & 15)) * C_ + kc * 32 + ((lane >> 4) << 4));
            v8h a0 = *(const v8h*)ab;
            v8h a1 = *(const v8h*)(ab + 8);
            v16h am = __builtin_shufflevector(a0, a1, 0,1,2,3,4,5,6,7,8,9,10,11,12,13,14,15);

            int c0 = kc * 32 + ((lane >> 4) << 4);     // B: 16 consecutive K per lane
            {
                int pl = (pt0)*16 + (lane & 15);
                const v8h* sp = (const v8h*)(&S[pl * SPITCH + c0]);
                v8h b0 = sp[0], b1 = sp[1];
                v16h bm = __builtin_shufflevector(b0, b1, 0,1,2,3,4,5,6,7,8,9,10,11,12,13,14,15);
                acc0 = __builtin_amdgcn_wmma_f32_16x16x32_f16(false, am, false, bm,
                                                              (short)0, acc0, false, false);
            }
            {
                int pl = (pt0 + 1)*16 + (lane & 15);
                const v8h* sp = (const v8h*)(&S[pl * SPITCH + c0]);
                v8h b0 = sp[0], b1 = sp[1];
                v16h bm = __builtin_shufflevector(b0, b1, 0,1,2,3,4,5,6,7,8,9,10,11,12,13,14,15);
                acc1 = __builtin_amdgcn_wmma_f32_16x16x32_f16(false, am, false, bm,
                                                              (short)0, acc1, false, false);
            }
        }
        __syncthreads();
    }

    // ---- store: D element (r, lane) -> o = obase + r, p = pt*16 + lane%16
#pragma unroll
    for (int r = 0; r < 8; ++r) {
        int o = obase + r;
        size_t rowb = ((size_t)b * O_ + o) * HW_ + pix0;
        out[rowb + (pt0)*16     + (lane & 15)] = acc0[r];
        out[rowb + (pt0 + 1)*16 + (lane & 15)] = acc1[r];
    }
}

// ---------------------------------------------------------------------------
extern "C" void kernel_launch(void* const* d_in, const int* in_sizes, int n_in,
                              void* d_out, int out_size, void* d_ws, size_t ws_size,
                              hipStream_t stream) {
    const float* x   = (const float*)d_in[0];
    const float* ow  = (const float*)d_in[1];
    const float* ob  = (const float*)d_in[2];
    const float* mw  = (const float*)d_in[3];
    const float* mb  = (const float*)d_in[4];
    const float* rw  = (const float*)d_in[5];
    const float* rb  = (const float*)d_in[6];
    float* out = (float*)d_out;

    // workspace layout (256B-aligned offsets), total ~23.9 MB
    char* ws = (char*)d_ws;
    _Float16* regw_h = (_Float16*)(ws);                       //  73,728 B
    float*    soff   = (float*)(ws + 73728);                  //  1,638,400 B
    float*    mbuf   = (float*)(ws + 1712128);                //  7,372,800 B
    float*    crd    = (float*)(ws + 9084928);                // 14,745,600 B

    k_convert_regw<<<(K_ * O_ * C_ + 255) / 256, 256, 0, stream>>>(rw, regw_h);
    k_small_conv  <<<(B_ * HW_) / 256, 256, 0, stream>>>(x, ow, ob, mw, mb, soff, mbuf);
    k_search      <<<(B_ * HW_) / 128, 128, 0, stream>>>(x, soff, crd);
    k_deform_wmma <<<B_ * (HW_ / 64), 256, 0, stream>>>(x, crd, mbuf, regw_h, rb, out);
}
